// MeanResidualMLPEnsemble_89137751261199
// MI455X (gfx1250) — matrix operand order repaired and checked by reading, not compiled, not run on hardware
//
#include <hip/hip_runtime.h>
#include <hip/hip_bf16.h>

// ---------------------------------------------------------------------------
// MeanResidualMLPEnsemble for MI455X (gfx1250, wave32, WMMA, async-LDS)
//
// Sizes: B=256, M=3, L=100000, A=4096, H=512, K1=M*A=12288
//
// Pipeline (all on `stream`):
//   1) transpose+f16-convert hidden_w  (H x K1) -> (K1 x H)         [ws]
//   2) transpose+f16-convert delta_layer_w (A x H) -> (H x A)       [ws]
//   3) gather x[:,:,active_idx] -> f16 x_active (B x K1)            [ws]
//   4) base logits: streaming einsum over x (memory-bound, ~410MB)  [d_out]
//   5) GEMM1 (WMMA f16, deterministic split-K=8) -> f32 partials    [ws]
//   6) reduce partials + bias + relu -> f16 h                       [ws]
//   7) GEMM2 (WMMA f16) + alpha-scaled atomic scatter into d_out
// ---------------------------------------------------------------------------

#define B_DIM   256
#define M_DIM   3
#define L_DIM   100000
#define A_DIM   4096
#define H_DIM   512
#define K1_DIM  (M_DIM * A_DIM)   // 12288
#define KSPLIT  8
#define KCHUNK  (K1_DIM / KSPLIT) // 1536

typedef __attribute__((ext_vector_type(16))) _Float16 v16h;
typedef __attribute__((ext_vector_type(8)))  _Float16 v8h;
typedef __attribute__((ext_vector_type(8)))  float    v8f;
typedef __attribute__((ext_vector_type(4)))  int      v4i;

#define AS_GLOBAL __attribute__((address_space(1)))
#define AS_SHARED __attribute__((address_space(3)))

#if __has_builtin(__builtin_amdgcn_global_load_async_to_lds_b128) && \
    __has_builtin(__builtin_amdgcn_s_wait_asynccnt)
#define HAVE_ASYNC_LDS 1
#else
#define HAVE_ASYNC_LDS 0
#endif

// ---------------- tiling parameters -----------------
#define TILE_M 64
#define TILE_N 64
#define TILE_K 32
#define A_LDS  40   // padded row stride (f16) for A tile: 80B, 16B aligned, conflict-free
#define B_LDS  72   // padded row stride (f16) for B tile: 144B, 16B aligned
#define A_TILE_H (TILE_M * A_LDS) // f16 elems per A buffer
#define B_TILE_H (TILE_K * B_LDS) // f16 elems per B buffer

// ---------------------------------------------------------------------------
// Fragment loaders (per CDNA5 ISA VGPR layouts, wave32)
// ---------------------------------------------------------------------------

// A-matrix 16x32 f16: lane t -> row = t&15; half = t>>4.
// Lane holds K = {half*8 + 0..7} and {16 + half*8 + 0..7}: two 16B chunks.
__device__ __forceinline__ v16h load_a_frag(const _Float16* As, int lane, int rbase) {
    const int row  = rbase + (lane & 15);
    const int half = (lane >> 4) & 1;
    const _Float16* p = As + row * A_LDS + half * 8;
    v8h lo = *(const v8h*)(p);        // K = half*8 + 0..7
    v8h hi = *(const v8h*)(p + 16);   // K = 16 + half*8 + 0..7
    union { v16h v; v8h h[2]; } u;
    u.h[0] = lo; u.h[1] = hi;
    return u.v;
}

// B-matrix 32x16 f16: lane t holds column N=t&15; K = {half*8 + 0..7} and {16 + half*8 + 0..7}.
__device__ __forceinline__ v16h load_b_frag(const _Float16* Bs, int lane, int ntile) {
    const int n    = ntile * 16 + (lane & 15);
    const int half = (lane >> 4) & 1;
    v16h v;
    #pragma unroll
    for (int s = 0; s < 8; ++s) {
        v[s]     = Bs[(half * 8 + s) * B_LDS + n];
        v[8 + s] = Bs[(16 + half * 8 + s) * B_LDS + n];
    }
    return v;
}

// ---------------------------------------------------------------------------
// Stage one 64x32 A tile + 32x64 B tile into LDS.
// Async path: GLOBAL_LOAD_ASYNC_TO_LDS_B128 (4 instructions/wave, ASYNCcnt).
// Builtin signature (from diagnostics): (AS1 int4*, AS3 int4*, imm off, imm cpol)
// ---------------------------------------------------------------------------
__device__ __forceinline__ void stage_tiles(
    const _Float16* __restrict__ Ag, size_t ldA,
    const _Float16* __restrict__ Bg, size_t ldB,
    int k0, _Float16* Asb, _Float16* Bsb, int tid)
{
#if HAVE_ASYNC_LDS
    #pragma unroll
    for (int i = 0; i < 2; ++i) {
        int c  = tid + i * 128;
        int r  = c >> 2;
        int co = (c & 3) * 8;
        __builtin_amdgcn_global_load_async_to_lds_b128(
            (AS_GLOBAL v4i*)(Ag + (size_t)r * ldA + k0 + co),
            (AS_SHARED v4i*)(Asb + r * A_LDS + co), 0, 0);
    }
    #pragma unroll
    for (int i = 0; i < 2; ++i) {
        int c  = tid + i * 128;
        int r  = c >> 3;
        int co = (c & 7) * 8;
        __builtin_amdgcn_global_load_async_to_lds_b128(
            (AS_GLOBAL v4i*)(Bg + (size_t)(k0 + r) * ldB + co),
            (AS_SHARED v4i*)(Bsb + r * B_LDS + co), 0, 0);
    }
#else
    #pragma unroll
    for (int i = 0; i < 2; ++i) {
        int c  = tid + i * 128;
        int r  = c >> 2;
        int co = (c & 3) * 8;
        *(v8h*)(Asb + r * A_LDS + co) = *(const v8h*)(Ag + (size_t)r * ldA + k0 + co);
    }
    #pragma unroll
    for (int i = 0; i < 2; ++i) {
        int c  = tid + i * 128;
        int r  = c >> 3;
        int co = (c & 7) * 8;
        *(v8h*)(Bsb + r * B_LDS + co) = *(const v8h*)(Bg + (size_t)(k0 + r) * ldB + co);
    }
#endif
}

// ---------------------------------------------------------------------------
// Double-buffered WMMA mainloop: one 64x64 C tile per 128-thread WG (4 waves).
// Wave w computes rows [w*16, w*16+16) x all 64 columns -> 4 accumulators.
// ---------------------------------------------------------------------------
__device__ __forceinline__ void gemm_tile_db(
    const _Float16* __restrict__ Ag, size_t ldA,
    const _Float16* __restrict__ Bg, size_t ldB,
    int K, _Float16* As, _Float16* Bs, int tid, v8f acc[4])
{
    const int lane = tid & 31;
    const int wave = tid >> 5;

    stage_tiles(Ag, ldA, Bg, ldB, 0, As, Bs, tid);

    int buf = 0;
    for (int k0 = 0; k0 < K; k0 += TILE_K) {
        _Float16* Ac = As + buf * A_TILE_H;
        _Float16* Bc = Bs + buf * B_TILE_H;
        if (k0 + TILE_K < K) {
            // Stage next slab into the other buffer while this one computes.
            stage_tiles(Ag, ldA, Bg, ldB, k0 + TILE_K,
                        As + (buf ^ 1) * A_TILE_H,
                        Bs + (buf ^ 1) * B_TILE_H, tid);
            // Prefetch the slab after next into L2.
            if (k0 + 2 * TILE_K < K) {
                __builtin_prefetch(Ag + (size_t)(tid >> 2) * ldA + k0 + 2 * TILE_K, 0, 1);
                __builtin_prefetch(Bg + (size_t)(k0 + 2 * TILE_K + (tid >> 3)) * ldB, 0, 1);
            }
#if HAVE_ASYNC_LDS
            __builtin_amdgcn_s_wait_asynccnt(4);   // current slab's 4 ops done
#endif
        } else {
#if HAVE_ASYNC_LDS
            __builtin_amdgcn_s_wait_asynccnt(0);
#endif
        }
        __syncthreads();   // all waves' staging of `buf` visible

        v16h af = load_a_frag(Ac, lane, wave * 16);
        v16h bf[4];
        #pragma unroll
        for (int nt = 0; nt < 4; ++nt) bf[nt] = load_b_frag(Bc, lane, nt);
        #pragma unroll
        for (int nt = 0; nt < 4; ++nt)
            acc[nt] = __builtin_amdgcn_wmma_f32_16x16x32_f16(
                false, af, false, bf[nt], (short)0, acc[nt], false, false);

        __syncthreads();   // protect `buf` before it is restaged next iteration
        buf ^= 1;
    }
}

// ---------------------------------------------------------------------------
// 1/2) LDS-tiled transpose + f32->f16 convert.
// src: R x C row-major.  dst[c*R + r] = (f16) src[r*C + c].  R,C multiples of 32.
// ---------------------------------------------------------------------------
__global__ void __launch_bounds__(256)
transpose_f32_to_f16(const float* __restrict__ src, _Float16* __restrict__ dst,
                     int R, int C)
{
    __shared__ float t[32][33];
    const int c0 = blockIdx.x * 32;
    const int r0 = blockIdx.y * 32;
    const int cl = threadIdx.x & 31;
    const int rl = threadIdx.x >> 5;   // 0..7
    #pragma unroll
    for (int j = 0; j < 4; ++j) {
        int r = rl + j * 8;
        t[cl][r] = src[(size_t)(r0 + r) * C + c0 + cl];   // coalesced read
    }
    __syncthreads();
    #pragma unroll
    for (int j = 0; j < 4; ++j) {
        int c = rl + j * 8;
        dst[(size_t)(c0 + c) * R + r0 + cl] = (_Float16)t[c][cl];  // coalesced write
    }
}

// ---------------------------------------------------------------------------
// 3) Gather x[:,:,active_idx] -> f16 x_active (B x K1), K1 index = m*A + a.
// ---------------------------------------------------------------------------
__global__ void __launch_bounds__(256)
gather_x_kernel(const float* __restrict__ x, const int* __restrict__ idx,
                _Float16* __restrict__ xa)
{
    int i = blockIdx.x * 256 + threadIdx.x;
    if (i >= B_DIM * K1_DIM) return;
    int b = i / K1_DIM;
    int r = i - b * K1_DIM;
    int m = r >> 12;          // / A_DIM
    int a = r & (A_DIM - 1);
    int l = idx[a];
    xa[i] = (_Float16)x[((size_t)b * M_DIM + m) * L_DIM + l];
}

// ---------------------------------------------------------------------------
// 4) Base logits (memory-bound streaming; float4):
//    out[b,l] = sum_m x[b,m,l] * (softmax(gl)[m] + delta_w[m,l]) + bias[l]
// ---------------------------------------------------------------------------
__global__ void __launch_bounds__(256)
base_logits_kernel(const float* __restrict__ x, const float* __restrict__ gl,
                   const float* __restrict__ dw, const float* __restrict__ bias,
                   float* __restrict__ out)
{
    int t = blockIdx.x * 256 + threadIdx.x;
    if (t >= L_DIM / 4) return;
    const int b = blockIdx.y;
    const int l = t * 4;

    float g0 = gl[0], g1 = gl[1], g2 = gl[2];
    float mx = fmaxf(g0, fmaxf(g1, g2));
    float e0 = expf(g0 - mx), e1 = expf(g1 - mx), e2 = expf(g2 - mx);
    float inv = 1.0f / (e0 + e1 + e2);
    float w0 = e0 * inv, w1 = e1 * inv, w2 = e2 * inv;

    const size_t xb = (size_t)b * M_DIM * L_DIM;
    float4 x0 = *(const float4*)(x + xb + 0 * (size_t)L_DIM + l);
    float4 x1 = *(const float4*)(x + xb + 1 * (size_t)L_DIM + l);
    float4 x2 = *(const float4*)(x + xb + 2 * (size_t)L_DIM + l);
    float4 d0 = *(const float4*)(dw + 0 * (size_t)L_DIM + l);
    float4 d1 = *(const float4*)(dw + 1 * (size_t)L_DIM + l);
    float4 d2 = *(const float4*)(dw + 2 * (size_t)L_DIM + l);
    float4 bb = *(const float4*)(bias + l);

    float4 o;
    o.x = fmaf(x0.x, w0 + d0.x, fmaf(x1.x, w1 + d1.x, fmaf(x2.x, w2 + d2.x, bb.x)));
    o.y = fmaf(x0.y, w0 + d0.y, fmaf(x1.y, w1 + d1.y, fmaf(x2.y, w2 + d2.y, bb.y)));
    o.z = fmaf(x0.z, w0 + d0.z, fmaf(x1.z, w1 + d1.z, fmaf(x2.z, w2 + d2.z, bb.z)));
    o.w = fmaf(x0.w, w0 + d0.w, fmaf(x1.w, w1 + d1.w, fmaf(x2.w, w2 + d2.w, bb.w)));
    *(float4*)(out + (size_t)b * L_DIM + l) = o;
}

// ---------------------------------------------------------------------------
// 5) GEMM1, deterministic split-K: partial[z] = xa[:, zK:(z+1)K] @ hwT[zK:(z+1)K, :]
//    grid (8, 4, 8) = 256 WGs, 48 K-steps each. No float atomics.
// ---------------------------------------------------------------------------
__global__ void __launch_bounds__(128)
gemm1_kernel(const _Float16* __restrict__ xa, const _Float16* __restrict__ hwT,
             float* __restrict__ partial)
{
    __shared__ __align__(16) _Float16 As[2 * A_TILE_H];
    __shared__ __align__(16) _Float16 Bs[2 * B_TILE_H];
    const int tid = threadIdx.x;
    const int wgN = blockIdx.x;   // 0..7   (512/64)
    const int wgM = blockIdx.y;   // 0..3   (256/64)
    const int wgK = blockIdx.z;   // 0..7   split-K

    v8f acc[4] = {};
    const _Float16* Ag = xa + (size_t)(wgM * TILE_M) * K1_DIM + (size_t)wgK * KCHUNK;
    const _Float16* Bg = hwT + (size_t)wgK * KCHUNK * H_DIM + wgN * TILE_N;
    gemm_tile_db(Ag, K1_DIM, Bg, H_DIM, KCHUNK, As, Bs, tid, acc);

    float* P = partial + (size_t)wgK * (B_DIM * H_DIM);
    const int lane = tid & 31;
    const int wave = tid >> 5;
    const int n_l  = lane & 15;
    const int mh   = (lane >> 4) * 8;
    #pragma unroll
    for (int nt = 0; nt < 4; ++nt) {
        int col = wgN * TILE_N + nt * 16 + n_l;
        #pragma unroll
        for (int v = 0; v < 8; ++v) {
            int row = wgM * TILE_M + wave * 16 + mh + v;
            P[(size_t)row * H_DIM + col] = acc[nt][v];
        }
    }
}

// ---------------------------------------------------------------------------
// 6) Reduce split-K partials + bias + relu -> f16 h (deterministic order).
// ---------------------------------------------------------------------------
__global__ void __launch_bounds__(256)
reduce_relu_kernel(const float* __restrict__ partial, const float* __restrict__ hidden_b,
                   _Float16* __restrict__ h)
{
    int i = blockIdx.x * 256 + threadIdx.x;     // < B*H = 131072
    int col = i & (H_DIM - 1);
    float s = 0.0f;
    #pragma unroll
    for (int k = 0; k < KSPLIT; ++k)
        s += partial[(size_t)k * (B_DIM * H_DIM) + i];
    h[i] = (_Float16)fmaxf(s + hidden_b[col], 0.0f);
}

// ---------------------------------------------------------------------------
// 7) GEMM2 + scatter: delta = h(256x512) @ dlwT(512x4096) + dlb;
//    out[b, active_idx[a]] += alpha * delta[b,a]  (f32 atomics; dups accumulate)
// ---------------------------------------------------------------------------
__global__ void __launch_bounds__(128)
gemm2_scatter_kernel(const _Float16* __restrict__ h, const _Float16* __restrict__ dlwT,
                     const float* __restrict__ dlb, const int* __restrict__ active_idx,
                     const float* __restrict__ log_alpha, float* __restrict__ out)
{
    __shared__ __align__(16) _Float16 As[2 * A_TILE_H];
    __shared__ __align__(16) _Float16 Bs[2 * B_TILE_H];
    const int tid = threadIdx.x;
    const int wgN = blockIdx.x;   // 0..63  (4096/64)
    const int wgM = blockIdx.y;   // 0..3

    v8f acc[4] = {};
    const _Float16* Ag = h + (size_t)(wgM * TILE_M) * H_DIM;
    const _Float16* Bg = dlwT + wgN * TILE_N;
    gemm_tile_db(Ag, H_DIM, Bg, A_DIM, H_DIM, As, Bs, tid, acc);

    const float alpha = 0.1f / (1.0f + expf(-log_alpha[0]));
    const int lane = tid & 31;
    const int wave = tid >> 5;
    const int n_l  = lane & 15;
    const int mh   = (lane >> 4) * 8;
    #pragma unroll
    for (int nt = 0; nt < 4; ++nt) {
        int a  = wgN * TILE_N + nt * 16 + n_l;
        float bv = dlb[a];
        int l  = active_idx[a];
        #pragma unroll
        for (int v = 0; v < 8; ++v) {
            int row = wgM * TILE_M + wave * 16 + mh + v;
            float val = alpha * (acc[nt][v] + bv);
            atomicAdd(out + (size_t)row * L_DIM + l, val);
        }
    }
}

// ---------------------------------------------------------------------------
// Launcher
// ---------------------------------------------------------------------------
extern "C" void kernel_launch(void* const* d_in, const int* in_sizes, int n_in,
                              void* d_out, int out_size, void* d_ws, size_t ws_size,
                              hipStream_t stream) {
    (void)in_sizes; (void)n_in; (void)out_size; (void)ws_size;

    const float* x          = (const float*)d_in[0];
    const int*   active_idx = (const int*)  d_in[1];
    const float* gl         = (const float*)d_in[2];
    const float* dw         = (const float*)d_in[3];
    const float* bias       = (const float*)d_in[4];
    const float* log_alpha  = (const float*)d_in[5];
    const float* hidden_w   = (const float*)d_in[6];
    const float* hidden_b   = (const float*)d_in[7];
    const float* dlw        = (const float*)d_in[8];
    const float* dlb        = (const float*)d_in[9];
    float* out = (float*)d_out;

    char* ws = (char*)d_ws;
    const size_t xa_bytes   = (size_t)B_DIM * K1_DIM * 2;           //  6,291,456
    const size_t hw_bytes   = (size_t)K1_DIM * H_DIM * 2;           // 12,582,912
    const size_t dlw_bytes  = (size_t)H_DIM * A_DIM * 2;            //  4,194,304
    const size_t h_bytes    = (size_t)B_DIM * H_DIM * 2;            //    262,144
    _Float16* xa      = (_Float16*)(ws);
    _Float16* hwT     = (_Float16*)(ws + xa_bytes);
    _Float16* dlwT    = (_Float16*)(ws + xa_bytes + hw_bytes);
    _Float16* hbuf    = (_Float16*)(ws + xa_bytes + hw_bytes + dlw_bytes);
    float*    partial = (float*)   (ws + xa_bytes + hw_bytes + dlw_bytes + h_bytes);
    // partial: KSPLIT * B * H * 4 = 4,194,304 B  (total ws ~27.5 MB)

    // Weight transposes + f16 converts.
    transpose_f32_to_f16<<<dim3(K1_DIM / 32, H_DIM / 32), 256, 0, stream>>>(
        hidden_w, hwT, H_DIM, K1_DIM);
    transpose_f32_to_f16<<<dim3(H_DIM / 32, A_DIM / 32), 256, 0, stream>>>(
        dlw, dlwT, A_DIM, H_DIM);

    // Gather active columns of x into f16 activation matrix.
    gather_x_kernel<<<(B_DIM * K1_DIM) / 256, 256, 0, stream>>>(x, active_idx, xa);

    // Streaming base logits (dominant memory cost, ~410 MB traffic -> ~18 us).
    base_logits_kernel<<<dim3((L_DIM / 4 + 255) / 256, B_DIM), 256, 0, stream>>>(
        x, gl, dw, bias, out);

    // WMMA GEMM1 with deterministic split-K (256 WGs), then reduce+relu.
    gemm1_kernel<<<dim3(H_DIM / TILE_N, B_DIM / TILE_M, KSPLIT), 128, 0, stream>>>(
        xa, hwT, partial);
    reduce_relu_kernel<<<(B_DIM * H_DIM) / 256, 256, 0, stream>>>(
        partial, hidden_b, hbuf);

    // WMMA GEMM2 + scaled atomic scatter into logits (256 WGs).
    gemm2_scatter_kernel<<<dim3(A_DIM / TILE_N, B_DIM / TILE_M), 128, 0, stream>>>(
        hbuf, dlwT, dlb, active_idx, log_alpha, out);
}
